// ChimeraMoEFFN_9234179686588
// MI455X (gfx1250) — matrix operand order repaired and checked by reading, not compiled, not run on hardware
//
#include <hip/hip_runtime.h>
#include <math.h>

typedef __attribute__((ext_vector_type(16))) __bf16 v16bf;
typedef __attribute__((ext_vector_type(8)))  float  v8f;
typedef unsigned short ushort_t;
typedef unsigned int   uint_t;

constexpr int T_    = 2048;   // B*S tokens
constexpr int D_    = 1024;
constexpr int E_    = 8;
constexpr int KTOP  = 2;
constexpr int F_    = 2048;
constexpr int SLOTS = T_ * KTOP;   // 4096 (token,expert) slots
constexpr int TILE  = 128;
constexpr int BK    = 32;
constexpr int A_STRIDE = 40;       // bf16 elems: 80B rows (16B aligned)
constexpr int B_STRIDE = 36;       // f32 elems: 144B rows (16B aligned)

#if __has_builtin(__builtin_amdgcn_global_load_async_to_lds_b128)
#define HAS_ASYNC 1
typedef __attribute__((ext_vector_type(4))) int v4i_t;
typedef __attribute__((address_space(1))) v4i_t* gv4_t;   // global source (AS1)
typedef __attribute__((address_space(3))) v4i_t* lv4_t;   // LDS destination (AS3)
#else
#define HAS_ASYNC 0
#endif

template <int N>
static __device__ __forceinline__ void wait_async() {
#if __has_builtin(__builtin_amdgcn_s_wait_asynccnt)
  __builtin_amdgcn_s_wait_asynccnt(N);
#else
  asm volatile("s_wait_asynccnt %0" ::"i"(N) : "memory");
#endif
}

// float -> bf16 (RNE) for stores
static __device__ __forceinline__ ushort_t f2bfu(float f) {
  uint_t u = __builtin_bit_cast(uint_t, f);
  u = (u + 0x7FFFu + ((u >> 16) & 1u)) >> 16;
  return (ushort_t)u;
}
static __device__ __forceinline__ __bf16 us2bf(ushort_t u) {
  return __builtin_bit_cast(__bf16, u);
}
static __device__ __forceinline__ void atomicAddF(float* p, float v) {
  __hip_atomic_fetch_add(p, v, __ATOMIC_RELAXED, __HIP_MEMORY_SCOPE_AGENT);
}
static __device__ __forceinline__ float fast_sigmoid(float x) {
  return __builtin_amdgcn_rcpf(1.f + __expf(-x));   // v_exp_f32 + v_rcp_f32
}

// ---------------------------------------------------------------- init
__global__ __launch_bounds__(256) void k_init(const float* __restrict__ x,
                                              float* __restrict__ out,
                                              int* cnt, int* fill, int* top1,
                                              float* sum_prob) {
  int i = blockIdx.x * 256 + threadIdx.x;
  if (i < T_ * D_) out[i] = x[i];           // out = x; GEMM2 accumulates on top
  if (blockIdx.x == 0 && threadIdx.x < E_) {
    cnt[threadIdx.x] = 0; fill[threadIdx.x] = 0;
    top1[threadIdx.x] = 0; sum_prob[threadIdx.x] = 0.f;
  }
}

// ---------------------------------------------------------------- router
// one wave32 per token: RMSNorm -> xn(bf16), router logits, softmax, top-2
__global__ __launch_bounds__(256) void k_router(const float* __restrict__ x,
                                                const float* __restrict__ gate_w,
                                                const float* __restrict__ norm_w,
                                                ushort_t* __restrict__ xnb,
                                                int* __restrict__ topk_e,
                                                float* __restrict__ topk_w,
                                                int* cnt, int* top1, float* sum_prob) {
  const int lane = threadIdx.x & 31;
  const int wave = threadIdx.x >> 5;
  const int t    = blockIdx.x * 8 + wave;
  const size_t tb = (size_t)t * D_;

  float xv[32];
  float ss = 0.f;
#pragma unroll
  for (int i = 0; i < 32; ++i) {
    int d = i * 32 + lane;
    xv[i] = x[tb + d];
    ss += xv[i] * xv[i];
  }
#pragma unroll
  for (int off = 16; off > 0; off >>= 1) ss += __shfl_xor(ss, off, 32);
  const float inv = rsqrtf(ss * (1.0f / D_) + 1.1920928955078125e-07f);

  float lg[E_];
#pragma unroll
  for (int e = 0; e < E_; ++e) lg[e] = 0.f;
#pragma unroll
  for (int i = 0; i < 32; ++i) {
    int d = i * 32 + lane;
    float xn = xv[i] * inv * norm_w[d];
    xnb[tb + d] = f2bfu(xn);
#pragma unroll
    for (int e = 0; e < E_; ++e) lg[e] += xn * gate_w[e * D_ + d];
  }
#pragma unroll
  for (int e = 0; e < E_; ++e)
#pragma unroll
    for (int off = 16; off > 0; off >>= 1) lg[e] += __shfl_xor(lg[e], off, 32);

  float m = lg[0];
#pragma unroll
  for (int e = 1; e < E_; ++e) m = fmaxf(m, lg[e]);
  float p[E_], s = 0.f;
#pragma unroll
  for (int e = 0; e < E_; ++e) { p[e] = __expf(lg[e] - m); s += p[e]; }
  float rs = 1.f / s;
#pragma unroll
  for (int e = 0; e < E_; ++e) p[e] *= rs;

  int i0 = 0;
#pragma unroll
  for (int e = 1; e < E_; ++e) if (p[e] > p[i0]) i0 = e;
  int i1 = (i0 == 0) ? 1 : 0;
#pragma unroll
  for (int e = 0; e < E_; ++e) if (e != i0 && p[e] > p[i1]) i1 = e;
  float wsum = fmaxf(p[i0] + p[i1], 1e-6f);
  float w0 = p[i0] / wsum, w1 = p[i1] / wsum;

  if (lane == 0) {
    topk_e[2 * t] = i0; topk_e[2 * t + 1] = i1;
    topk_w[2 * t] = w0; topk_w[2 * t + 1] = w1;
    atomicAdd(&cnt[i0], 1); atomicAdd(&cnt[i1], 1);
    atomicAdd(&top1[i0], 1);
  }
  if (lane < E_) atomicAddF(&sum_prob[lane], p[lane]);
}

// ---------------------------------------------------------------- scan + lb loss
__global__ void k_scan(const int* cnt, int* offs, int* fill,
                       const int* top1, const float* sum_prob,
                       float* __restrict__ out) {
  int o = 0;
  float lb = 0.f;
  for (int e = 0; e < E_; ++e) {
    offs[e] = o; fill[e] = o; o += cnt[e];
    lb += ((float)top1[e] / T_) * (sum_prob[e] / T_);
  }
  offs[E_] = o;
  out[(size_t)T_ * D_] = (float)E_ * lb;
}

// ---------------------------------------------------------------- compaction
__global__ __launch_bounds__(256) void k_place(const int* __restrict__ topk_e,
                                               const float* __restrict__ topk_w,
                                               int* fill,
                                               int* __restrict__ slot_token,
                                               float* __restrict__ slot_w) {
  int t = blockIdx.x * 256 + threadIdx.x;
  if (t >= T_) return;
#pragma unroll
  for (int k = 0; k < KTOP; ++k) {
    int e = topk_e[2 * t + k];
    int pos = atomicAdd(&fill[e], 1);
    slot_token[pos] = t;
    slot_w[pos] = topk_w[2 * t + k];
  }
}

// ---------------------------------------------------------------- fused expert GEMM
// MODE 0: h = silu(xn[gathered] @ w1[e]^T)   A: xnb (bf16, gathered rows), B: w1 (f32), K=D
// MODE 1: out += sc*slot_w*(h @ w2[e]^T)     A: hbuf (bf16, compacted),    B: w2 (f32), K=F
template <int MODE>
__global__ __launch_bounds__(256) void k_gemm(const ushort_t* __restrict__ Aglob,
                                              const float* __restrict__ Bglob,
                                              const int* __restrict__ slot_token,
                                              const float* __restrict__ slot_w,
                                              const int* __restrict__ offs,
                                              const float* __restrict__ scale,
                                              ushort_t* __restrict__ hbuf,
                                              float* __restrict__ out) {
  constexpr int KDIM = (MODE == 0) ? D_ : F_;
  constexpr int NW   = (MODE == 0) ? F_ : D_;   // B rows per expert

  const int e  = blockIdx.x >> 4;
  const int mt = blockIdx.x & 15;
  const int nt = blockIdx.y;
  const int base = offs[e];
  const int ne   = offs[e + 1] - base;
  if (mt * TILE >= ne) return;                  // block-uniform: EXEC all-ones at WMMA

  __shared__ __align__(16) ushort_t As[2][TILE][A_STRIDE];
  __shared__ __align__(16) float    Bsf[2][TILE][B_STRIDE];
  __shared__ int   rowTok[TILE];
  __shared__ float rowW[TILE];

  const int tid  = threadIdx.x;
  const int lane = tid & 31, wave = tid >> 5;
  const int wm = wave >> 1, wn = wave & 1;
  const int half = lane >> 4, l16 = lane & 15;

  if (tid < TILE) {
    int r = mt * TILE + tid; if (r > ne - 1) r = ne - 1;
    rowTok[tid] = slot_token[base + r];
    rowW[tid]   = slot_w[base + r];
  }
  __syncthreads();

  const float* Bbase = Bglob + ((size_t)e * NW + (size_t)nt * TILE) * KDIM;

  // per-thread staging coordinates (fixed across k-steps)
  int arow[2], akc[2];
#pragma unroll
  for (int r = 0; r < 2; ++r) { int idx = tid + r * 256; arow[r] = idx >> 2; akc[r] = (idx & 3) * 8; }
  int brow[4], bkc[4];
#pragma unroll
  for (int r = 0; r < 4; ++r) { int idx = tid + r * 256; brow[r] = idx >> 3; bkc[r] = (idx & 7) * 4; }

  size_t aoff[2];
#pragma unroll
  for (int r = 0; r < 2; ++r) {
    if (MODE == 0) {
      aoff[r] = (size_t)rowTok[arow[r]] * KDIM;                 // gather tokens
    } else {
      int ar = mt * TILE + arow[r]; if (ar > ne - 1) ar = ne - 1;
      aoff[r] = (size_t)(base + ar) * KDIM;                     // compacted h rows
    }
  }

  auto stage = [&](int buf, int k0) {
#pragma unroll
    for (int r = 0; r < 2; ++r) {
      const ushort_t* g = Aglob + aoff[r] + k0 + akc[r];
#if HAS_ASYNC
      __builtin_amdgcn_global_load_async_to_lds_b128((gv4_t)g, (lv4_t)&As[buf][arow[r]][akc[r]], 0, 0);
#else
      *(uint4*)&As[buf][arow[r]][akc[r]] = *(const uint4*)g;
#endif
    }
#pragma unroll
    for (int r = 0; r < 4; ++r) {
      const float* g = Bbase + (size_t)brow[r] * KDIM + k0 + bkc[r];
#if HAS_ASYNC
      __builtin_amdgcn_global_load_async_to_lds_b128((gv4_t)g, (lv4_t)&Bsf[buf][brow[r]][bkc[r]], 0, 0);
#else
      *(float4*)&Bsf[buf][brow[r]][bkc[r]] = *(const float4*)g;
#endif
    }
  };

  v8f acc[2][4];
#pragma unroll
  for (int a = 0; a < 2; ++a)
#pragma unroll
    for (int b = 0; b < 4; ++b)
#pragma unroll
      for (int i = 0; i < 8; ++i) acc[a][b][i] = 0.f;

  stage(0, 0);                                   // prologue

  constexpr int NK = KDIM / BK;
  for (int kt = 0; kt < NK; ++kt) {
    const int cur = kt & 1;
    if (kt + 1 < NK) stage(cur ^ 1, (kt + 1) * BK);   // prefetch next tile (overlaps WMMA below)
#if HAS_ASYNC
    if (kt + 1 < NK) wait_async<6>();            // my 6 cur-tile copies done; 6 next in flight
    else             wait_async<0>();
#endif
    __syncthreads();                             // whole WG's cur tile visible

    v16bf afr[2];
#pragma unroll
    for (int a = 0; a < 2; ++a) {
      int mrow = wm * 32 + a * 16 + l16;
#pragma unroll
      for (int j = 0; j < 8; ++j) {
        int kk = 2 * j + ((j >= 4) ? 8 : 0) + 8 * half;   // ISA 16-bit A layout
        uint_t u = *(const uint_t*)&As[cur][mrow][kk];
        afr[a][2 * j]     = us2bf((ushort_t)(u & 0xffffu));
        afr[a][2 * j + 1] = us2bf((ushort_t)(u >> 16));
      }
    }
    v16bf bfr[4];
#pragma unroll
    for (int b = 0; b < 4; ++b) {
      int nrow = wn * 64 + b * 16 + l16;
#pragma unroll
      for (int j = 0; j < 8; ++j) {
        int kk = 2 * j + ((j >= 4) ? 8 : 0) + 8 * half;   // ISA 16-bit B layout (same k order as A)
        float2 q = *(const float2*)&Bsf[cur][nrow][kk];
        bfr[b][2 * j]     = (__bf16)q.x;                  // cvt co-executes with WMMA
        bfr[b][2 * j + 1] = (__bf16)q.y;
      }
    }
#pragma unroll
    for (int a = 0; a < 2; ++a)
#pragma unroll
      for (int b = 0; b < 4; ++b)
        acc[a][b] = __builtin_amdgcn_wmma_f32_16x16x32_bf16(
            false, afr[a], false, bfr[b], (short)0, acc[a][b], false, false);
    __syncthreads();                             // protect cur LDS from next overwrite
  }

  const float sc = (MODE == 1) ? fast_sigmoid(scale[0]) : 0.f;
#pragma unroll
  for (int a = 0; a < 2; ++a)
#pragma unroll
    for (int b = 0; b < 4; ++b)
#pragma unroll
      for (int i = 0; i < 8; ++i) {
        int ml = wm * 32 + a * 16 + half * 8 + i;         // C layout: VGPR i -> M = i + 8*(lane/16)
        int gm = mt * TILE + ml;
        if (gm < ne) {
          int nl = nt * TILE + wn * 64 + b * 16 + l16;
          float v = acc[a][b][i];
          if (MODE == 0) {
            float sv = v * fast_sigmoid(v);               // silu
            hbuf[(size_t)(base + gm) * F_ + nl] = f2bfu(sv);
          } else {
            int tok = rowTok[ml];
            atomicAddF(&out[(size_t)tok * D_ + nl], sc * rowW[ml] * v);
          }
        }
      }
}

// ---------------------------------------------------------------- launch
extern "C" void kernel_launch(void* const* d_in, const int* in_sizes, int n_in,
                              void* d_out, int out_size, void* d_ws, size_t ws_size,
                              hipStream_t stream) {
  const float* x      = (const float*)d_in[0];  // [B,S,D]
  const float* gate_w = (const float*)d_in[1];  // [E,D]
  const float* w1     = (const float*)d_in[2];  // [E,F,D]
  const float* w2     = (const float*)d_in[3];  // [E,D,F]
  const float* norm_w = (const float*)d_in[4];  // [D]
  const float* scale  = (const float*)d_in[5];  // [1]
  float* out = (float*)d_out;                   // [T*D] + 1 (lb loss)

  char* base = (char*)d_ws;
  size_t o = 0;
  auto carve = [&](size_t bytes) -> char* {
    o = (o + 255) & ~(size_t)255;
    char* p = base + o; o += bytes; return p;
  };
  ushort_t* xnb        = (ushort_t*)carve((size_t)T_ * D_ * 2);    // 4 MB
  ushort_t* hbuf       = (ushort_t*)carve((size_t)SLOTS * F_ * 2); // 16 MB
  int*      slot_token = (int*)carve(SLOTS * 4);
  float*    slot_w     = (float*)carve(SLOTS * 4);
  int*      topk_e     = (int*)carve(T_ * KTOP * 4);
  float*    topk_w     = (float*)carve(T_ * KTOP * 4);
  int*      cnt        = (int*)carve(E_ * 4);
  int*      offs       = (int*)carve((E_ + 1) * 4);
  int*      fill       = (int*)carve(E_ * 4);
  int*      top1       = (int*)carve(E_ * 4);
  float*    sum_prob   = (float*)carve(E_ * 4);
  (void)ws_size; (void)in_sizes; (void)n_in; (void)out_size;

  k_init  <<<(T_ * D_ + 255) / 256, 256, 0, stream>>>(x, out, cnt, fill, top1, sum_prob);
  k_router<<<T_ / 8, 256, 0, stream>>>(x, gate_w, norm_w, xnb, topk_e, topk_w, cnt, top1, sum_prob);
  k_scan  <<<1, 1, 0, stream>>>(cnt, offs, fill, top1, sum_prob, out);
  k_place <<<T_ / 256, 256, 0, stream>>>(topk_e, topk_w, fill, slot_token, slot_w);
  k_gemm<0><<<dim3(E_ * 16, F_ / TILE), 256, 0, stream>>>(xnb, w1, slot_token, slot_w, offs, scale, hbuf, out);
  k_gemm<1><<<dim3(E_ * 16, D_ / TILE), 256, 0, stream>>>(hbuf, w2, slot_token, slot_w, offs, scale, hbuf, out);
}
